// AttentionBlock_996432413488
// MI455X (gfx1250) — matrix operand order repaired and checked by reading, not compiled, not run on hardware
//
#include <hip/hip_runtime.h>

// ---------- types & helpers ----------
typedef __attribute__((ext_vector_type(16))) __bf16        v16bf;
typedef __attribute__((ext_vector_type(8)))  float         v8f;
typedef __attribute__((ext_vector_type(4)))  unsigned int  v4u;
typedef __attribute__((ext_vector_type(8)))  int           v8i;

union FragAB { v16bf bf; v4u u4[2]; };

__device__ inline unsigned short f2bf(float f) {
  unsigned int u = __float_as_uint(f);
  u += 0x7FFFu + ((u >> 16) & 1u);           // round-to-nearest-even
  return (unsigned short)(u >> 16);
}

// A fragment: 16x32 bf16, source row-major (rows = M, stride ld elems).
// lane<16: row l, k = [0..7] & [16..23]; lane>=16: row l, k = [8..15] & [24..31]
__device__ inline v16bf load_frag_a(const unsigned short* base, int ld, int lane) {
  FragAB f;
  int half = lane >> 4, l = lane & 15;
  const unsigned short* p = base + (size_t)l * ld + (half ? 8 : 0);
  f.u4[0] = *(const v4u*)(p);
  f.u4[1] = *(const v4u*)(p + 16);
  return f.bf;
}

// B fragment: 32x16 bf16, source = B^T row-major (rows = N cols, stride ld).
// lane<16: col l, k = [0..7] & [16..23]; lane>=16: col l, k = [8..15] & [24..31]
__device__ inline v16bf load_frag_b(const unsigned short* base, int ld, int lane) {
  FragAB f;
  int half = lane >> 4, l = lane & 15;
  const unsigned short* p = base + (size_t)l * ld + (half ? 8 : 0);
  f.u4[0] = *(const v4u*)(p);
  f.u4[1] = *(const v4u*)(p + 16);
  return f.bf;
}

__device__ inline v8f wmma_bf16(v16bf a, v16bf b, v8f c) {
  return __builtin_amdgcn_wmma_f32_16x16x32_bf16(false, a, false, b, (short)0, c, false, false);
}

__device__ inline void wait_tensorcnt0() {
#if __has_builtin(__builtin_amdgcn_s_wait_tensorcnt)
  __builtin_amdgcn_s_wait_tensorcnt(0);
#else
  asm volatile("s_wait_tensorcnt 0x0" ::: "memory");
#endif
}

// TDM: DMA a 2D bf16 tile (tile_d1 rows x tile_d0 elems, row stride stride0 elems)
// from global memory into LDS (packed row-major). D# per ISA 08_async_tensor §8.
__device__ inline void tdm_load_2d(unsigned int ldsOff, const void* gptr,
                                   unsigned int tensor_d0, unsigned int tensor_d1,
                                   unsigned int tile_d0,   unsigned int tile_d1,
                                   unsigned long long stride0) {
  unsigned long long ga = (unsigned long long)gptr;
  v4u g0;
  g0[0] = 1u;                                            // count=1, user descriptor
  g0[1] = ldsOff;                                        // lds_addr (bytes)
  g0[2] = (unsigned int)(ga & 0xFFFFFFFFu);              // global_addr[31:0]
  g0[3] = (unsigned int)((ga >> 32) & 0x01FFFFFFu)       // global_addr[56:32]
        | (2u << 30);                                    // type = 2 (image)
  v8i g1;
  g1[0] = (int)(1u << 16);                               // data_size=1 (2B), mask=0
  g1[1] = (int)((tensor_d0 & 0xFFFFu) << 16);            // tensor_dim0[15:0]
  g1[2] = (int)((tensor_d0 >> 16) | ((tensor_d1 & 0xFFFFu) << 16)); // d0[31:16], d1[15:0]
  g1[3] = (int)((tensor_d1 >> 16) | (tile_d0 << 16));    // d1[31:16], tile_dim0
  g1[4] = (int)(tile_d1 & 0xFFFFu);                      // tile_dim1 (tile_dim2=0)
  g1[5] = (int)(stride0 & 0xFFFFFFFFull);                // tensor_dim0_stride[31:0]
  g1[6] = (int)((stride0 >> 32) & 0xFFFFull);            // stride[47:32], dim1_stride=0
  g1[7] = 0;
  asm volatile("tensor_load_to_lds %0, %1" :: "s"(g0), "s"(g1) : "memory");
}

__device__ inline unsigned int lds_off(const void* p) {
  // generic pointer to LDS: low 32 bits are the LDS byte offset (ISA §10.2 aperture)
  return (unsigned int)(unsigned long long)p;
}

// ---------- constants ----------
#define BATCH 4
#define CCH   256
#define NPIX  4096     // 64*64
#define NHEAD 4
#define HDIM  64
#define GRPS  8
#define GC    32       // channels per group

// ---------- kernel 1: GroupNorm statistics ----------
__global__ void gn_stats_kernel(const float* __restrict__ x, float* __restrict__ stats) {
  int bg = blockIdx.x;                       // 0..31 == b*8+g
  const int CNT = GC * NPIX;                 // 131072
  size_t base = (size_t)bg * CNT;            // groups are contiguous in C
  float s = 0.f, ss = 0.f;
  for (int i = threadIdx.x; i < CNT; i += blockDim.x) {
    float v = x[base + i];
    s += v; ss += v * v;
  }
  __shared__ float r1[256], r2[256];
  r1[threadIdx.x] = s; r2[threadIdx.x] = ss;
  __syncthreads();
  for (int off = 128; off > 0; off >>= 1) {
    if ((int)threadIdx.x < off) { r1[threadIdx.x] += r1[threadIdx.x + off];
                                  r2[threadIdx.x] += r2[threadIdx.x + off]; }
    __syncthreads();
  }
  if (threadIdx.x == 0) {
    float mean = r1[0] / (float)CNT;
    float var  = r2[0] / (float)CNT - mean * mean;
    stats[bg * 2]     = mean;
    stats[bg * 2 + 1] = rsqrtf(var + 1e-5f);
  }
}

// ---------- kernel 2: GN apply + transpose to channel-last bf16 h[b][n][c] ----------
__global__ void gn_apply_kernel(const float* __restrict__ x, const float* __restrict__ gw,
                                const float* __restrict__ gb, const float* __restrict__ stats,
                                unsigned short* __restrict__ h) {
  int id = blockIdx.x;               // 4 * 8 * 64 = 2048 blocks
  int nt = id & 63, ct = (id >> 6) & 7, b = id >> 9;
  int c0 = ct * GC, n0 = nt * 64;
  __shared__ float tile[32][65];
  float mean = stats[(b * GRPS + ct) * 2];
  float rstd = stats[(b * GRPS + ct) * 2 + 1];
  for (int i = threadIdx.x; i < 32 * 64; i += 256) {
    int nn = i & 63, cc = i >> 6;
    float v = x[((size_t)(b * CCH + c0 + cc)) * NPIX + n0 + nn];
    tile[cc][nn] = (v - mean) * rstd * gw[c0 + cc] + gb[c0 + cc];
  }
  __syncthreads();
  for (int i = threadIdx.x; i < 32 * 64; i += 256) {
    int cc = i & 31, nn = i >> 5;
    h[((size_t)(b * NPIX + n0 + nn)) * CCH + c0 + cc] = f2bf(tile[cc][nn]);
  }
}

// ---------- kernel 3: weight fp32 -> bf16 ----------
__global__ void cvt_w_kernel(const float* __restrict__ wq, const float* __restrict__ wp,
                             unsigned short* __restrict__ wq_b, unsigned short* __restrict__ wp_b) {
  int i = blockIdx.x * 256 + threadIdx.x;
  if (i < 3 * CCH * CCH) wq_b[i] = f2bf(wq[i]);
  if (i < CCH * CCH)     wp_b[i] = f2bf(wp[i]);
}

// ---------- kernel 4: QKV GEMM, scatter into q[n][d], k[m][d], v[d][m] ----------
__global__ void __launch_bounds__(256)
qkv_gemm_kernel(const unsigned short* __restrict__ w, const float* __restrict__ bias,
                const unsigned short* __restrict__ h,
                unsigned short* __restrict__ qT, unsigned short* __restrict__ kT,
                unsigned short* __restrict__ vB) {
  int lane = threadIdx.x & 31, wid = threadIdx.x >> 5;
  int gwv = blockIdx.x * 8 + wid;            // 0..12287  (4 * 48 * 64)
  int b = gwv / (48 * 64);
  int rem = gwv % (48 * 64);
  int o0 = (rem / 64) * 16, n0 = (rem % 64) * 64;
  int half = lane >> 4, lcol = lane & 15;
  v8f acc[4] = {};
  for (int c0 = 0; c0 < CCH; c0 += 32) {
    v16bf a = load_frag_a(w + (size_t)o0 * CCH + c0, CCH, lane);
#pragma unroll
    for (int t = 0; t < 4; ++t) {
      v16bf bb = load_frag_b(h + ((size_t)b * NPIX + n0 + t * 16) * CCH + c0, CCH, lane);
      acc[t] = wmma_bf16(a, bb, acc[t]);
    }
  }
#pragma unroll
  for (int t = 0; t < 4; ++t) {
#pragma unroll
    for (int r = 0; r < 8; ++r) {
      int o = o0 + r + half * 8;
      int n = n0 + t * 16 + lcol;
      unsigned short bv = f2bf(acc[t][r] + bias[o]);
      int seg = o >> 8;                       // 0=q 1=k 2=v
      int oc  = o & 255;
      int hd  = oc >> 6, d = oc & 63;
      size_t bh = (size_t)(b * NHEAD + hd);
      if (seg == 0)      qT[(bh * NPIX + n) * HDIM + d] = bv;
      else if (seg == 1) kT[(bh * NPIX + n) * HDIM + d] = bv;
      else               vB[(bh * HDIM + d) * NPIX + n] = bv;
    }
  }
}

// ---------- kernel 5: flash attention (TDM double-buffered tiles, all-WMMA) ----------
#define BC 64
__global__ void __launch_bounds__(256)
attn_kernel(const unsigned short* __restrict__ qT, const unsigned short* __restrict__ kT,
            const unsigned short* __restrict__ vB, unsigned short* __restrict__ oT) {
  int lane = threadIdx.x & 31, wid = threadIdx.x >> 5;
  int half = lane >> 4, lcol = lane & 15;
  int blk = blockIdx.x;                       // 512 = 16 (b,h) * 32 row-blocks
  int rb = blk & 31, bh = blk >> 5;
  int b = bh >> 2, hh = bh & 3;
  int n_base = rb * 128 + wid * 16;

  const unsigned short* qbase = qT + ((size_t)bh * NPIX + n_base) * HDIM;
  const unsigned short* kbase = kT + (size_t)bh * NPIX * HDIM;
  const unsigned short* vbase = vB + (size_t)bh * HDIM * NPIX;

  v16bf qf0 = load_frag_a(qbase + 0,  HDIM, lane);   // d 0..31
  v16bf qf1 = load_frag_a(qbase + 32, HDIM, lane);   // d 32..63

  __shared__ __align__(16) unsigned short k_lds[2][BC * HDIM];   // [m][d], double-buffered
  __shared__ __align__(16) unsigned short v_lds[2][HDIM * BC];   // [d][m], double-buffered
  __shared__ __align__(16) unsigned short p_lds[8][16 * BC];     // per-wave P staging

  v8f o_acc[4] = {};
  float m_r[8], l_r[8];
#pragma unroll
  for (int r = 0; r < 8; ++r) { m_r[r] = -1e30f; l_r[r] = 0.f; }
  const float scale = 0.125f;                 // 1/sqrt(64)

  // preload first tile pair via Tensor Data Mover (wave 0 issues, TENSORcnt tracks)
  if (wid == 0) {
    tdm_load_2d(lds_off(k_lds[0]), kbase, BC * HDIM, 1, BC * HDIM, 1, BC * HDIM);
    tdm_load_2d(lds_off(v_lds[0]), vbase, NPIX, HDIM, BC, HDIM, NPIX);
    wait_tensorcnt0();
  }
  __syncthreads();

  int pb = 0;
  for (int m0 = 0; m0 < NPIX; m0 += BC) {
    // kick off DMA of the next K/V tile into the other buffer (overlaps compute)
    if (wid == 0 && m0 + BC < NPIX) {
      tdm_load_2d(lds_off(k_lds[pb ^ 1]), kbase + (size_t)(m0 + BC) * HDIM,
                  BC * HDIM, 1, BC * HDIM, 1, BC * HDIM);
      tdm_load_2d(lds_off(v_lds[pb ^ 1]), vbase + (m0 + BC),
                  NPIX, HDIM, BC, HDIM, NPIX);
    }
    const unsigned short* kc = k_lds[pb];
    const unsigned short* vc = v_lds[pb];

    // S = Q K^T  (4 col-tiles of 16 keys)
    v8f s[4];
#pragma unroll
    for (int t = 0; t < 4; ++t) {
      v16bf b0 = load_frag_b(kc + (t * 16) * HDIM + 0,  HDIM, lane);
      v16bf b1 = load_frag_b(kc + (t * 16) * HDIM + 32, HDIM, lane);
      v8f c = {};
      c = wmma_bf16(qf0, b0, c);
      c = wmma_bf16(qf1, b1, c);
      s[t] = c;
    }

    // online softmax (rows live in fragment slot r, 16-lane halves)
    float mnew[8];
#pragma unroll
    for (int r = 0; r < 8; ++r) {
      float mx = -1e30f;
#pragma unroll
      for (int t = 0; t < 4; ++t) { s[t][r] *= scale; mx = fmaxf(mx, s[t][r]); }
#pragma unroll
      for (int off = 8; off >= 1; off >>= 1) mx = fmaxf(mx, __shfl_xor(mx, off, 32));
      mnew[r] = fmaxf(m_r[r], mx);
    }
#pragma unroll
    for (int r = 0; r < 8; ++r) {
      float sum = 0.f;
#pragma unroll
      for (int t = 0; t < 4; ++t) {
        float p = __expf(s[t][r] - mnew[r]);
        s[t][r] = p; sum += p;
      }
#pragma unroll
      for (int off = 8; off >= 1; off >>= 1) sum += __shfl_xor(sum, off, 32);
      float alpha = __expf(m_r[r] - mnew[r]);
      l_r[r] = l_r[r] * alpha + sum;
      m_r[r] = mnew[r];
#pragma unroll
      for (int t = 0; t < 4; ++t) o_acc[t][r] *= alpha;
    }

    // P: D-layout -> A-layout via per-wave LDS staging (same-wave LDS is in-order)
    unsigned short* pw = p_lds[wid];
#pragma unroll
    for (int t = 0; t < 4; ++t)
#pragma unroll
      for (int r = 0; r < 8; ++r)
        pw[(r + half * 8) * BC + t * 16 + lcol] = f2bf(s[t][r]);

    // O += P V^T  (4 d-tiles of 16)
    v16bf pa0 = load_frag_a(pw + 0,  BC, lane);       // m 0..31
    v16bf pa1 = load_frag_a(pw + 32, BC, lane);       // m 32..63
#pragma unroll
    for (int t = 0; t < 4; ++t) {
      v16bf b0 = load_frag_b(vc + (t * 16) * BC + 0,  BC, lane);
      v16bf b1 = load_frag_b(vc + (t * 16) * BC + 32, BC, lane);
      o_acc[t] = wmma_bf16(pa0, b0, o_acc[t]);
      o_acc[t] = wmma_bf16(pa1, b1, o_acc[t]);
    }

    // wave 0 drains the DMA; barrier releases everyone onto the fresh buffer
    if (wid == 0) wait_tensorcnt0();
    __syncthreads();
    pb ^= 1;
  }

  // epilogue: O /= l, write channel-last oT[b][n][h*64+d] (bf16)
#pragma unroll
  for (int r = 0; r < 8; ++r) {
    float inv = 1.f / l_r[r];
    int n = n_base + r + half * 8;
#pragma unroll
    for (int t = 0; t < 4; ++t) {
      int c = hh * HDIM + t * 16 + lcol;
      oT[((size_t)b * NPIX + n) * CCH + c] = f2bf(o_acc[t][r] * inv);
    }
  }
}

// ---------- kernel 6: proj GEMM + bias + residual (fp32 out) ----------
__global__ void __launch_bounds__(256)
proj_gemm_kernel(const unsigned short* __restrict__ w, const float* __restrict__ bias,
                 const unsigned short* __restrict__ oT, const float* __restrict__ x,
                 float* __restrict__ out) {
  int lane = threadIdx.x & 31, wid = threadIdx.x >> 5;
  int gwv = blockIdx.x * 8 + wid;            // 0..4095 (4 * 16 * 64)
  int b = gwv / (16 * 64);
  int rem = gwv % (16 * 64);
  int o0 = (rem / 64) * 16, n0 = (rem % 64) * 64;
  int half = lane >> 4, lcol = lane & 15;
  v8f acc[4] = {};
  for (int c0 = 0; c0 < CCH; c0 += 32) {
    v16bf a = load_frag_a(w + (size_t)o0 * CCH + c0, CCH, lane);
#pragma unroll
    for (int t = 0; t < 4; ++t) {
      v16bf bb = load_frag_b(oT + ((size_t)b * NPIX + n0 + t * 16) * CCH + c0, CCH, lane);
      acc[t] = wmma_bf16(a, bb, acc[t]);
    }
  }
#pragma unroll
  for (int t = 0; t < 4; ++t)
#pragma unroll
    for (int r = 0; r < 8; ++r) {
      int o = o0 + r + half * 8;
      int n = n0 + t * 16 + lcol;
      size_t idx = ((size_t)(b * CCH + o)) * NPIX + n;
      out[idx] = x[idx] + acc[t][r] + bias[o];
    }
}

// ---------- launch ----------
extern "C" void kernel_launch(void* const* d_in, const int* in_sizes, int n_in,
                              void* d_out, int out_size, void* d_ws, size_t ws_size,
                              hipStream_t stream) {
  const float* x      = (const float*)d_in[0];
  const float* gn_w   = (const float*)d_in[1];
  const float* gn_b   = (const float*)d_in[2];
  const float* qkv_w  = (const float*)d_in[3];
  const float* qkv_b  = (const float*)d_in[4];
  const float* proj_w = (const float*)d_in[5];
  const float* proj_b = (const float*)d_in[6];
  float* out = (float*)d_out;

  char* ws = (char*)d_ws;
  float*          stats = (float*)ws;                                   //     256 B
  unsigned short* wq    = (unsigned short*)(ws + 512);                  //  384 KB
  unsigned short* wp    = (unsigned short*)(ws + 512 + 393216);         //  128 KB
  unsigned short* h     = (unsigned short*)(ws + 512 + 393216 + 131072);//    8 MB
  unsigned short* qT    = (unsigned short*)((char*)h  + 8388608);
  unsigned short* kT    = (unsigned short*)((char*)qT + 8388608);
  unsigned short* vB    = (unsigned short*)((char*)kT + 8388608);
  unsigned short* oT    = (unsigned short*)((char*)vB + 8388608);

  gn_stats_kernel <<<  32, 256, 0, stream>>>(x, stats);
  cvt_w_kernel    <<< 768, 256, 0, stream>>>(qkv_w, proj_w, wq, wp);
  gn_apply_kernel <<<2048, 256, 0, stream>>>(x, gn_w, gn_b, stats, h);
  qkv_gemm_kernel <<<1536, 256, 0, stream>>>(wq, qkv_b, h, qT, kT, vB);
  attn_kernel     <<< 512, 256, 0, stream>>>(qT, kT, vB, oT);
  proj_gemm_kernel<<< 512, 256, 0, stream>>>(wp, proj_b, oT, x, out);
}
